// WindowHeadAttention_59210419143191
// MI455X (gfx1250) — matrix operand order repaired and checked by reading, not compiled, not run on hardware
//
#include <hip/hip_runtime.h>

// WindowHeadAttention fused kernel for gfx1250 (MI455X).
// bf16 WMMA GEMMs (v_wmma_f32_16x16x32_bf16) + fused circular window-5 sum,
// online softmax over n, weighted reduction. Global->LDS staging uses
// CDNA5 async loads (global_load_async_to_lds_b128 / ASYNCcnt) double-buffered.

typedef __bf16 bf16_t;
typedef __attribute__((ext_vector_type(16))) __bf16 v16bf;
typedef __attribute__((ext_vector_type(8)))  __bf16 v8bf;
typedef __attribute__((ext_vector_type(8)))  float  v8f;

#define NSEQ  4096
#define DDIM  512
#define CHUNK 64          // d-columns per workgroup
#define NWAVE 4           // waves per workgroup (wave32)
#define NTHR  (NWAVE * 32)
#define KP    520         // padded row stride (bf16 elems) for LDS bf16 tiles

// ---- LDS layout (byte offsets) ----
#define OFF_WKT   0                         // [80][KP] bf16  Wk^T slice (A, ctx)
#define OFF_WVT   (OFF_WKT  + 80*KP*2)      // [64][KP] bf16  Wv^T slice (A, values)
#define OFF_KEYT  (OFF_WVT  + 64*KP*2)      // [16][KP] bf16  current key tile (B)
#define OFF_VALT  (OFF_KEYT + 16*KP*2)      // [16][KP] bf16  current val tile (B)
#define OFF_QS    (OFF_VALT + 16*KP*2)      // [80] f32       q = query@Wq slice
#define OFF_STAGE (OFF_QS   + 80*4)         // 2 x (16x512 f32 key + 16x512 f32 val)
#define STAGE_BUF_BYTES (16*512*4*2)        // 64 KB per buffer (key+val)
#define SMEM_BYTES (OFF_STAGE + 2*STAGE_BUF_BYTES)   // 314,432 B < 320 KB

__device__ __forceinline__ void async_b128(const void* g, unsigned ldsaddr) {
    unsigned long long ga = (unsigned long long)(uintptr_t)g;
    asm volatile("global_load_async_to_lds_b128 %0, %1, off"
                 :: "v"(ldsaddr), "v"(ga) : "memory");
}
__device__ __forceinline__ void wait_async0() {
    asm volatile("s_wait_asynccnt 0x0" ::: "memory");
}
__device__ __forceinline__ v16bf ld_v16(const bf16_t* p) {
    v8bf lo = *(const v8bf*)p;
    v8bf hi = *(const v8bf*)(p + 8);
    return __builtin_shufflevector(lo, hi, 0,1,2,3,4,5,6,7,8,9,10,11,12,13,14,15);
}

__global__ __launch_bounds__(NTHR)
void wha_fused_kernel(const float* __restrict__ query,
                      const float* __restrict__ key,
                      const float* __restrict__ val,
                      const float* __restrict__ Wq,
                      const float* __restrict__ Wk,
                      const float* __restrict__ Wv,
                      float* __restrict__ out)
{
    extern __shared__ char smem[];
    bf16_t* WkT  = (bf16_t*)(smem + OFF_WKT);
    bf16_t* WvT  = (bf16_t*)(smem + OFF_WVT);
    bf16_t* keyT = (bf16_t*)(smem + OFF_KEYT);
    bf16_t* valT = (bf16_t*)(smem + OFF_VALT);
    float*  qs   = (float*) (smem + OFF_QS);

    const int tid  = threadIdx.x;
    const int lane = tid & 31;
    const int wid  = tid >> 5;            // 0..3  (16-wide d sub-chunk)
    const int half = (lane >> 4) & 1;     // C-layout lane half (M rows 0-7 / 8-15)
    const int lm   = lane & 15;           // n-column within tile / A row
    const int b    = blockIdx.y;
    const int d0   = blockIdx.x * CHUNK;

    // ---- prologue: issue async staging of tile 0 (raw f32) into stage buf 0 ----
    {
        const float* krow = key + ((size_t)b * NSEQ) * DDIM;
        const float* vrow = val + ((size_t)b * NSEQ) * DDIM;
        unsigned sk = (unsigned)(uintptr_t)(smem + OFF_STAGE);
        unsigned sv = sk + 16*512*4;
        for (int idx = tid; idx < 2048; idx += NTHR) {          // 2048 16B chunks
            int n = idx >> 7, k4 = (idx & 127) << 2;
            async_b128(krow + n*DDIM + k4, sk + (unsigned)(n*512 + k4)*4u);
            async_b128(vrow + n*DDIM + k4, sv + (unsigned)(n*512 + k4)*4u);
        }
    }

    // ---- one-time staging: Wk^T slice (80 cols, circular), Wv^T slice (64 cols) ----
    for (int idx = tid; idx < 80 * DDIM; idx += NTHR) {
        int k = idx / 80, c = idx % 80;                 // coalesced over c
        int cg = (d0 + c) & (DDIM - 1);
        WkT[c * KP + k] = (bf16_t)Wk[k * DDIM + cg];
    }
    for (int idx = tid; idx < 64 * DDIM; idx += NTHR) {
        int k = idx / 64, c = idx % 64;
        WvT[c * KP + k] = (bf16_t)Wv[k * DDIM + d0 + c];
    }
    // q slice: qs[c] = sum_k query[b,k] * Wq[k,(d0+c)%512]  (tiny: 80x512 MACs)
    if (tid < 80) {
        int cg = (d0 + tid) & (DDIM - 1);
        float s = 0.f;
        for (int k = 0; k < DDIM; ++k)
            s += query[b * DDIM + k] * Wq[k * DDIM + cg];
        qs[tid] = s;
    }
    __syncthreads();

    // per-lane q constants: this lane's 8 d's are c = 16*wid + j + 8*half
    float q0[8], q1[4];
    for (int j = 0; j < 8; ++j) q0[j] = qs[16 * wid + j + 8 * half];
    for (int j = 0; j < 4; ++j) q1[j] = qs[16 * wid + 16 + j + 8 * half];

    // online-softmax state: 8 independent (m,l,acc) per lane (one per owned d)
    float m[8], l[8], a[8];
    for (int j = 0; j < 8; ++j) { m[j] = -3.0e38f; l[j] = 0.f; a[j] = 0.f; }

    const float RS5 = 0.4472135954999579f;   // 1/sqrt(WINDOW)
    const int ah = half * 8;                 // A-layout K sub-offset per lane half
    const int bh = half * 16;                // unused; B loads use ah, same K layout as A

    for (int it = 0; it < NSEQ / 16; ++it) {
        const int cur = it & 1;
        // tile `it` finished landing in stage[cur]
        wait_async0();
        __syncthreads();

        // issue async staging of tile it+1 into the other buffer (overlaps below)
        if (it + 1 < NSEQ / 16) {
            const int nb1 = (it + 1) * 16;
            const float* krow = key + ((size_t)b * NSEQ + nb1) * DDIM;
            const float* vrow = val + ((size_t)b * NSEQ + nb1) * DDIM;
            unsigned sk = (unsigned)(uintptr_t)(smem + OFF_STAGE + (cur ^ 1) * STAGE_BUF_BYTES);
            unsigned sv = sk + 16*512*4;
            for (int idx = tid; idx < 2048; idx += NTHR) {
                int n = idx >> 7, k4 = (idx & 127) << 2;
                async_b128(krow + n*DDIM + k4, sk + (unsigned)(n*512 + k4)*4u);
                async_b128(vrow + n*DDIM + k4, sv + (unsigned)(n*512 + k4)*4u);
            }
        }

        // ---- convert staged f32 tile -> bf16 operand tiles (LDS -> LDS) ----
        {
            const float* stK = (const float*)(smem + OFF_STAGE + cur * STAGE_BUF_BYTES);
            const float* stV = stK + 16*512;
            for (int idx = tid; idx < 2048; idx += NTHR) {
                int n = idx >> 7, k4 = (idx & 127) << 2;
                const float4 kf = *(const float4*)(stK + n*512 + k4);
                const float4 vf = *(const float4*)(stV + n*512 + k4);
                bf16_t* kd = &keyT[n * KP + k4];
                bf16_t* vd = &valT[n * KP + k4];
                kd[0] = (bf16_t)kf.x; kd[1] = (bf16_t)kf.y; kd[2] = (bf16_t)kf.z; kd[3] = (bf16_t)kf.w;
                vd[0] = (bf16_t)vf.x; vd[1] = (bf16_t)vf.y; vd[2] = (bf16_t)vf.z; vd[3] = (bf16_t)vf.w;
            }
        }
        __syncthreads();

        // ---- 3 transposed GEMM tiles: C[m=d][n], K=512 in 16 steps of 32 ----
        v8f c0 = {}; v8f c1 = {}; v8f cv = {};
        for (int kk = 0; kk < 16; ++kk) {
            const int kb = kk * 32;
            // B operand: column n = lm; same K layout as A -> K {0-7,16-23} / {8-15,24-31}
            v16bf bk = __builtin_shufflevector(*(const v8bf*)&keyT[lm * KP + kb + ah], *(const v8bf*)&keyT[lm * KP + kb + 16 + ah], 0,1,2,3,4,5,6,7,8,9,10,11,12,13,14,15);
            v16bf bv = __builtin_shufflevector(*(const v8bf*)&valT[lm * KP + kb + ah], *(const v8bf*)&valT[lm * KP + kb + 16 + ah], 0,1,2,3,4,5,6,7,8,9,10,11,12,13,14,15);
            // A operand: row m = lm; ISA 16-bit A layout -> K {0-7,16-23} / {8-15,24-31}
            const bf16_t* ar0 = &WkT[(16 * wid      + lm) * KP + kb];
            const bf16_t* ar1 = &WkT[(16 * wid + 16 + lm) * KP + kb];
            const bf16_t* arv = &WvT[(16 * wid      + lm) * KP + kb];
            v8bf a0l = *(const v8bf*)(ar0 + ah);
            v8bf a0h = *(const v8bf*)(ar0 + 16 + ah);
            v8bf a1l = *(const v8bf*)(ar1 + ah);
            v8bf a1h = *(const v8bf*)(ar1 + 16 + ah);
            v8bf avl = *(const v8bf*)(arv + ah);
            v8bf avh = *(const v8bf*)(arv + 16 + ah);
            v16bf a0 = __builtin_shufflevector(a0l, a0h, 0,1,2,3,4,5,6,7,8,9,10,11,12,13,14,15);
            v16bf a1 = __builtin_shufflevector(a1l, a1h, 0,1,2,3,4,5,6,7,8,9,10,11,12,13,14,15);
            v16bf av = __builtin_shufflevector(avl, avh, 0,1,2,3,4,5,6,7,8,9,10,11,12,13,14,15);
            c0 = __builtin_amdgcn_wmma_f32_16x16x32_bf16(false, a0, false, bk, (short)0, c0, false, false);
            c1 = __builtin_amdgcn_wmma_f32_16x16x32_bf16(false, a1, false, bk, (short)0, c1, false, false);
            cv = __builtin_amdgcn_wmma_f32_16x16x32_bf16(false, av, false, bv, (short)0, cv, false, false);
        }
        // NOTE: keyT/valT are re-written only after next loop-top barrier.

        // ---- prods + circular window-5 sum over d (lane-local after 2 shuffles) ----
        float own[8], oth[8], ext[4];
        for (int j = 0; j < 8; ++j) own[j] = q0[j] * c0[j];
        for (int j = 0; j < 8; ++j) oth[j] = __shfl_xor(own[j], 16, 32);
        for (int j = 0; j < 4; ++j) {
            float p1 = q1[j] * c1[j];            // tile1 cols d0w+16..19 live in lower half
            ext[j] = __shfl(p1, lm, 32);         // broadcast lower-half value to both halves
        }
        float pr[20];
        for (int j = 0; j < 8; ++j) {
            pr[j]     = half ? oth[j] : own[j];
            pr[8 + j] = half ? own[j] : oth[j];
        }
        for (int j = 0; j < 4; ++j) pr[16 + j] = ext[j];

        // ---- online softmax over n + weighted accumulation of values ----
        for (int j = 0; j < 8; ++j) {
            const int base = 8 * half + j;
            float w  = (pr[base] + pr[base+1] + pr[base+2] + pr[base+3] + pr[base+4]) * RS5;
            float vv = cv[j];
            float mn = fmaxf(m[j], w);
            float sc = __expf(m[j] - mn);
            float p  = __expf(w - mn);
            l[j] = l[j] * sc + p;
            a[j] = a[j] * sc + p * vv;
            m[j] = mn;
        }
    }

    // ---- merge the 16 per-lane partial softmaxes sharing each d-set ----
    for (int s = 1; s < 16; s <<= 1) {
        for (int j = 0; j < 8; ++j) {
            float mo = __shfl_xor(m[j], s, 32);
            float lo = __shfl_xor(l[j], s, 32);
            float ao = __shfl_xor(a[j], s, 32);
            float mn = fmaxf(m[j], mo);
            float e0 = __expf(m[j] - mn), e1 = __expf(mo - mn);
            l[j] = l[j] * e0 + lo * e1;
            a[j] = a[j] * e0 + ao * e1;
            m[j] = mn;
        }
    }
    if (lm == 0) {
        for (int j = 0; j < 8; ++j)
            out[b * DDIM + d0 + 16 * wid + 8 * half + j] = a[j] / l[j];
    }
}

extern "C" void kernel_launch(void* const* d_in, const int* in_sizes, int n_in,
                              void* d_out, int out_size, void* d_ws, size_t ws_size,
                              hipStream_t stream) {
    (void)in_sizes; (void)n_in; (void)out_size; (void)d_ws; (void)ws_size;
    const float* query = (const float*)d_in[0];
    const float* key   = (const float*)d_in[1];
    const float* val   = (const float*)d_in[2];
    const float* Wq    = (const float*)d_in[3];
    const float* Wk    = (const float*)d_in[4];
    const float* Wv    = (const float*)d_in[5];
    float* out = (float*)d_out;

    dim3 grid(DDIM / CHUNK, 32, 1);    // 8 d-chunks x 32 batches = 256 WGs
    dim3 block(NTHR, 1, 1);            // 4 waves (wave32)
    wha_fused_kernel<<<grid, block, SMEM_BYTES, stream>>>(query, key, val, Wq, Wk, Wv, out);
}
